// MetaGraphSci_59244778881421
// MI455X (gfx1250) — compile-verified, matrix operand order checked by reading, not run on hardware
//
#include <hip/hip_runtime.h>
#include <hip/hip_bf16.h>
#include <math.h>

// ---------------------------------------------------------------------------
// MetaGraphSci pipeline for gfx1250 (MI455X). All heavy GEMMs + attention use
// v_wmma_f32_16x16x32_f16 (wave32 WMMA). LDS tiles are stored in a
// fragment-ready permuted layout (k-groups [0..7|16..23|8..15|24..31] per
// 32-chunk) so each lane's WMMA operand is two aligned ds_load_b128.
// GEMM: register-double-buffered (next tile's global loads overlap WMMA),
// gather templated out, bounds-check-free (all launches tile-aligned:
// M%64==0, N%64==0, K%64==0). Workspace requirement: ~210 MB.
// ---------------------------------------------------------------------------

typedef __attribute__((ext_vector_type(16))) _Float16 v16h;
typedef __attribute__((ext_vector_type(8)))  _Float16 v8h;   // 16B
typedef __attribute__((ext_vector_type(4)))  _Float16 h4;    // 8B
typedef __attribute__((ext_vector_type(8)))  float    v8f;
typedef __attribute__((ext_vector_type(4)))  float    v4f;

#define NEGBIG (-1e9f)

__device__ inline float gelu_f(float x) {
    return 0.5f * x * (1.0f + erff(x * 0.70710678118654752f));
}
__device__ inline float wred_add(float v) {
    #pragma unroll
    for (int o = 16; o > 0; o >>= 1) v += __shfl_xor(v, o, 32);
    return v;
}
// permuted position of k within a 32-chunk: group g=k>>3 -> bit-reversed(g).
// 4-aligned runs of 4 stay contiguous (they never cross an 8-group).
__device__ inline int permk(int k) {
    int g = (k >> 3) & 3;
    int pg = ((g & 1) << 1) | (g >> 1);
    return (pg << 3) | (k & 7);
}
// full permuted position within a 64-wide row (two 32-chunks)
__device__ inline int permk64(int k) {
    return (k & 32) + permk(k & 31);
}
// load one 16-half fragment (32B) from permuted LDS row, 16B-aligned
__device__ inline v16h frag_ld(const _Float16* p) {
    v8h lo = *(const v8h*)p;
    v8h hi = *(const v8h*)(p + 8);
    return __builtin_shufflevector(lo, hi, 0,1,2,3,4,5,6,7,8,9,10,11,12,13,14,15);
}
__device__ inline h4 pack4(v4f a) {
    h4 r = { (_Float16)a.x, (_Float16)a.y, (_Float16)a.z, (_Float16)a.w };
    return r;
}

// ---------------------------------------------------------------------------
// WMMA GEMM: C[M,N] = op(A_gathered[M,K] @ B[K,N])
// flags: 1=+bias[n], 2=gelu, 4=+res[m,n].  GATHER: A row m -> gather[m].
// Tile 64x64x64, 128 threads (4 waves); wave w computes rows [16w,16w+16).
// REQUIRES: M%64==0, N%64==0, K%64==0 (true for all call sites).
// LDS rows: 72 halves (144B); register double-buffered global fetch.
// ---------------------------------------------------------------------------
template<bool GATHER>
__global__ __launch_bounds__(128)
void gemm_wmma_f16(const float* __restrict__ A, const float* __restrict__ B,
                   const float* __restrict__ bias, const float* __restrict__ res,
                   float* __restrict__ C, const int* __restrict__ gather,
                   int M, int N, int K, int flags)
{
    __shared__ __align__(16) _Float16 As[64 * 72];
    __shared__ __align__(16) _Float16 Bs[64 * 72];
    const int tid = threadIdx.x, wave = tid >> 5, lane = tid & 31;
    const int bn = blockIdx.x * 64, bm = blockIdx.y * 64;
    const int l15 = lane & 15, lhi = (lane >> 4) & 1;

    // loop-invariant staging pointers / LDS offsets (gather resolved once)
    const float* aptr[8]; int aoff[8];
    const float* bptr[8]; int boff[8];
    #pragma unroll
    for (int s = 0; s < 8; s++) {
        int i = tid + s * 128;                  // [0, 1024)
        int m = i >> 4, kq = (i & 15) << 2;
        int ar = GATHER ? gather[bm + m] : (bm + m);
        aptr[s] = A + (size_t)ar * K + kq;
        aoff[s] = m * 72 + permk64(kq);
        int n = i & 63, kqb = (i >> 6) << 2;
        bptr[s] = B + (size_t)kqb * N + bn + n;
        boff[s] = n * 72 + permk64(kqb);
    }

    v4f aReg[8], bReg[8];
    auto fetch = [&](int k0) {
        #pragma unroll
        for (int s = 0; s < 8; s++) aReg[s] = *(const v4f*)(aptr[s] + k0);
        #pragma unroll
        for (int s = 0; s < 8; s++) {
            const float* p = bptr[s] + (size_t)k0 * N;
            v4f t = { p[0], p[(size_t)N], p[(size_t)2 * N], p[(size_t)3 * N] };
            bReg[s] = t;
        }
    };

    v8f acc[4] = {};
    fetch(0);
    for (int k0 = 0; k0 < K; k0 += 64) {
        #pragma unroll
        for (int s = 0; s < 8; s++) *(h4*)&As[aoff[s]] = pack4(aReg[s]);
        #pragma unroll
        for (int s = 0; s < 8; s++) *(h4*)&Bs[boff[s]] = pack4(bReg[s]);
        __syncthreads();

        if (k0 + 64 < K) fetch(k0 + 64);        // overlap with WMMAs below

        #pragma unroll
        for (int ks = 0; ks < 2; ks++) {
            v16h af = frag_ld(&As[(wave * 16 + l15) * 72 + ks * 32 + lhi * 16]);
            #pragma unroll
            for (int c = 0; c < 4; c++) {
                v16h bf = frag_ld(&Bs[(c * 16 + l15) * 72 + ks * 32 + lhi * 16]);
                acc[c] = __builtin_amdgcn_wmma_f32_16x16x32_f16(
                    false, af, false, bf, (short)0, acc[c], false, false);
            }
        }
        __syncthreads();
    }

    // epilogue: D layout — lane&15 = col, vgpr v -> row v + 8*(lane>=16)
    #pragma unroll
    for (int c = 0; c < 4; c++) {
        int n = bn + c * 16 + l15;
        float bv = (flags & 1) ? bias[n] : 0.f;
        #pragma unroll
        for (int v = 0; v < 8; v++) {
            int m = bm + wave * 16 + lhi * 8 + v;
            float x = acc[c][v] + bv;
            if (flags & 2) x = gelu_f(x);
            if (flags & 4) x += res[(size_t)m * N + n];
            C[(size_t)m * N + n] = x;
        }
    }
}

// ---------------------------------------------------------------------------
// Small scalar GEMM for selector precomputes:
// C[m,n] = sum_k A[m*K+k] * B[(bRow0+k)*ldb + n]  (+ bias[n])
// ---------------------------------------------------------------------------
__global__ void naive_gemm(const float* __restrict__ A, const float* __restrict__ B,
                           const float* __restrict__ bias, float* __restrict__ C,
                           int M, int N, int K, int ldb, int bRow0)
{
    int idx = blockIdx.x * blockDim.x + threadIdx.x;
    if (idx >= M * N) return;
    int m = idx / N, n = idx % N;
    float s = bias ? bias[n] : 0.f;
    for (int k = 0; k < K; k++) s += A[m * K + k] * B[(size_t)(bRow0 + k) * ldb + n];
    C[idx] = s;
}

// ---------------------------------------------------------------------------
// Selector logits: one wave per (b,n). Cosine similarity + folded f1 blocks +
// exact GELU + f2 dot.
// ---------------------------------------------------------------------------
__global__ __launch_bounds__(128)
void selector_logits(const float* __restrict__ center, const float* __restrict__ cand,
                     const unsigned char* __restrict__ cmask,
                     const int* __restrict__ etyp, const float* __restrict__ ydel,
                     const float* __restrict__ cscr,
                     const float* __restrict__ Hc, const float* __restrict__ basev,
                     const float* __restrict__ edgev,
                     const float* __restrict__ tvec, const float* __restrict__ tconst,
                     const float* __restrict__ svec, const float* __restrict__ sconst,
                     const float* __restrict__ cconst,
                     const float* __restrict__ f1w, const float* __restrict__ f1b,
                     const float* __restrict__ f2w, const float* __restrict__ f2b,
                     float* __restrict__ logits)
{
    const int wave = threadIdx.x >> 5, lane = threadIdx.x & 31;
    const int row = blockIdx.x * 4 + wave;      // [0, 32*1024)
    const int b = row >> 10;

    const float* ct = center + (size_t)b * 768;
    const float* ce = cand + (size_t)row * 768;
    float dot = 0.f, nc = 0.f, nn = 0.f;
    for (int i = lane; i < 768; i += 32) {
        float a = ct[i], c = ce[i];
        dot += a * c; nc += a * a; nn += c * c;
    }
    dot = wred_add(dot); nc = wred_add(nc); nn = wred_add(nn);
    float sem = dot / (fmaxf(sqrtf(nc), 1e-8f) * fmaxf(sqrtf(nn), 1e-8f));

    int et = etyp[row]; if (et < 0) et = 0;
    float yd = ydel[row], cs = cscr[row];
    const float* semrow = f1w + (size_t)1280 * 256;

    float part = 0.f;
    for (int j = lane; j < 256; j += 32) {
        float f1 = Hc[(size_t)row * 256 + j] + basev[b * 256 + j] + edgev[et * 256 + j]
                 + yd * tvec[j] + tconst[j] + cs * svec[j] + sconst[j] + cconst[j]
                 + sem * semrow[j] + f1b[j];
        part += gelu_f(f1) * f2w[j];
    }
    part = wred_add(part);
    if (lane == 0) {
        float lg = part + f2b[0];
        if (!cmask[row]) lg = NEGBIG;
        logits[row] = lg;
    }
}

// ---------------------------------------------------------------------------
// Top-256 of 1024 per batch: bitonic sort (desc, tie-break by low index).
// ---------------------------------------------------------------------------
__global__ __launch_bounds__(512)
void topk1024(const float* __restrict__ logits, float* __restrict__ topv,
              int* __restrict__ topi)
{
    __shared__ float v[1024];
    __shared__ int ix[1024];
    const int b = blockIdx.x, t = threadIdx.x;
    for (int i = t; i < 1024; i += 512) { v[i] = logits[b * 1024 + i]; ix[i] = i; }
    __syncthreads();
    for (int k = 2; k <= 1024; k <<= 1) {
        for (int j = k >> 1; j > 0; j >>= 1) {
            for (int i = t; i < 1024; i += 512) {
                int p = i ^ j;
                if (p > i) {
                    float va = v[i], vb = v[p];
                    int ia = ix[i], ib = ix[p];
                    bool aGreater = (va > vb) || (va == vb && ia < ib);
                    bool desc = ((i & k) == 0);
                    if (desc ? !aGreater : aGreater) {
                        v[i] = vb; v[p] = va; ix[i] = ib; ix[p] = ia;
                    }
                }
            }
            __syncthreads();
        }
    }
    if (t < 256) { topv[b * 256 + t] = v[t]; topi[b * 256 + t] = ix[t]; }
}

__global__ void build_rows(const int* __restrict__ topi, int* __restrict__ rows)
{
    int i = blockIdx.x * 256 + threadIdx.x;
    if (i < 32 * 256) rows[i] = (i >> 8) * 1024 + topi[i];
}

// ---------------------------------------------------------------------------
// Tokenizer struct terms + LayerNorm (one wave per token row, D=512)
// ---------------------------------------------------------------------------
__global__ __launch_bounds__(128)
void tok_struct_ln(const float* __restrict__ textp, const int* __restrict__ topi,
                   const float* __restrict__ topv, const int* __restrict__ etyp,
                   const float* __restrict__ ydel,
                   const float* __restrict__ edge_emb,
                   const float* __restrict__ t_w, const float* __restrict__ t_b,
                   const float* __restrict__ s_w, const float* __restrict__ s_b,
                   const float* __restrict__ hop_b, const float* __restrict__ sp_b,
                   const float* __restrict__ ln_g, const float* __restrict__ ln_b,
                   float* __restrict__ tokens)
{
    const int wave = threadIdx.x >> 5, lane = threadIdx.x & 31;
    const int row = blockIdx.x * 4 + wave;       // [0, 8192)
    const int b = row >> 8;
    int idx = topi[row];
    int et = etyp[b * 1024 + idx]; if (et < 0) et = 0;
    float yd = ydel[b * 1024 + idx];
    float tl = topv[row];

    const float* tp = textp + (size_t)row * 512;
    float* out = tokens + (size_t)row * 512;
    float s = 0.f, s2 = 0.f;
    for (int d = lane; d < 512; d += 32) {
        float x = tp[d] + edge_emb[et * 512 + d] + yd * t_w[d] + t_b[d]
                + tl * s_w[d] + s_b[d] + hop_b[d] + sp_b[d];
        out[d] = x; s += x; s2 += x * x;
    }
    s = wred_add(s); s2 = wred_add(s2);
    float mean = s * (1.f / 512.f);
    float var = s2 * (1.f / 512.f) - mean * mean;
    float r = rsqrtf(var + 1e-5f);
    for (int d = lane; d < 512; d += 32)
        out[d] = (out[d] - mean) * r * ln_g[d] + ln_b[d];
}

// ---------------------------------------------------------------------------
// LayerNorm over D=512, one wave per row
// ---------------------------------------------------------------------------
__global__ __launch_bounds__(128)
void ln512(const float* __restrict__ in, float* __restrict__ out,
           const float* __restrict__ g, const float* __restrict__ bb, int rows)
{
    const int wave = threadIdx.x >> 5, lane = threadIdx.x & 31;
    const int row = blockIdx.x * 4 + wave;
    if (row >= rows) return;
    const float* x = in + (size_t)row * 512;
    float s = 0.f, s2 = 0.f;
    for (int d = lane; d < 512; d += 32) { float v = x[d]; s += v; s2 += v * v; }
    s = wred_add(s); s2 = wred_add(s2);
    float mean = s * (1.f / 512.f);
    float var = s2 * (1.f / 512.f) - mean * mean;
    float r = rsqrtf(var + 1e-5f);
    float* y = out + (size_t)row * 512;
    for (int d = lane; d < 512; d += 32) y[d] = (x[d] - mean) * r * g[d] + bb[d];
}

// ---------------------------------------------------------------------------
// Attention: one block = (b, h, 16-query tile). 128 threads / 4 waves.
// Phases: [stage Q,K] -> [S=scale*QK^T WMMA] -> [V restage into K's region,
// overlapped with bias pass] -> [softmax -> permuted f16 P] -> [P@V WMMA].
// LDS arena (63KB):
//   Sc : f32[16][256]       @ 0      (16384B)
//   red: f32[16][8]         @ 16384  (512B)
//   Qs : f16[16*72]         @ 16896  (2304B)   row stride 144B
//   Ps : f16[16*264]        @ 19200  (8448B)   row stride 528B
//   KV : f16[256*72]        @ 27648  (36864B)  K rows 144B; V' rows 528B
// ---------------------------------------------------------------------------
__global__ __launch_bounds__(128)
void attn256(const float* __restrict__ Qb, const float* __restrict__ Kb,
             const float* __restrict__ Vb,
             const float* __restrict__ ab, const float* __restrict__ mc,
             const unsigned char* __restrict__ kpm,
             const float* __restrict__ sbw, const float* __restrict__ sbb,
             const float* __restrict__ mbw, const float* __restrict__ mbb,
             float* __restrict__ Out)
{
    __shared__ __align__(16) unsigned char smem[64512];
    float*    Sc  = (float*)smem;                   // [16][256]
    float*    red = (float*)(smem + 16384);         // [16][8]
    _Float16* Qs  = (_Float16*)(smem + 16896);      // [16*72]
    _Float16* Ps  = (_Float16*)(smem + 19200);      // [16*264]
    _Float16* KV  = (_Float16*)(smem + 27648);      // K:[256*72] / V':[64*264]

    const int blk = blockIdx.x;
    const int qt = blk & 15, h = (blk >> 4) & 7, b = blk >> 7;
    const int tid = threadIdx.x, wave = tid >> 5, lane = tid & 31;
    const int l15 = lane & 15, lhi = (lane >> 4) & 1;
    const int q0 = qt * 16;

    // stage Q (16x64): float4 -> h4 -> b64 store, permuted
    #pragma unroll
    for (int s = 0; s < 2; s++) {
        int i = tid + s * 128;                       // [0, 256)
        int r = i >> 4, kq = (i & 15) << 2;
        v4f q = *(const v4f*)(Qb + ((size_t)(b * 256 + q0 + r)) * 512 + h * 64 + kq);
        *(h4*)&Qs[r * 72 + permk64(kq)] = pack4(q);
    }
    // stage K (256 keys x 64 dh), permuted
    #pragma unroll
    for (int s = 0; s < 32; s++) {
        int i = tid + s * 128;                       // [0, 4096)
        int key = i >> 4, kq = (i & 15) << 2;
        v4f k = *(const v4f*)(Kb + ((size_t)(b * 256 + key)) * 512 + h * 64 + kq);
        *(h4*)&KV[key * 72 + permk64(kq)] = pack4(k);
    }
    __syncthreads();

    // --- S = Q K^T * scale ---
    for (int t = 0; t < 4; t++) {
        int kb = (wave + 4 * t) * 16;
        v8f acc = {};
        #pragma unroll
        for (int kc = 0; kc < 64; kc += 32) {
            v16h af = frag_ld(&Qs[l15 * 72 + kc + lhi * 16]);
            v16h bf = frag_ld(&KV[(kb + l15) * 72 + kc + lhi * 16]);
            acc = __builtin_amdgcn_wmma_f32_16x16x32_f16(
                false, af, false, bf, (short)0, acc, false, false);
        }
        #pragma unroll
        for (int v = 0; v < 8; v++)
            Sc[(v + lhi * 8) * 256 + kb + l15] = acc[v] * 0.125f;  // 64^-0.5
    }
    __syncthreads();   // S in LDS; K region now dead

    // --- stage V' (dh-major, permuted keys) into KV; overlaps bias pass ---
    #pragma unroll
    for (int s = 0; s < 32; s++) {
        int i = tid + s * 128;                        // [0, 4096)
        int dh = i & 63, kq = (i >> 6) << 2;          // 4 consecutive keys
        const float* src = Vb + ((size_t)(b * 256 + kq)) * 512 + h * 64 + dh;
        v4f t = { src[0], src[512], src[1024], src[1536] };
        *(h4*)&KV[dh * 264 + permk64(kq & 63) + (kq & 192)] = pack4(t);
    }
    // --- learned biases + key padding mask (touches Sc only) ---
    #pragma unroll
    for (int s = 0; s < 32; s++) {
        int i = tid + s * 128;
        int m = i >> 8, n = i & 255;
        size_t qk = ((size_t)(b * 256 + q0 + m)) * 256 + n;
        float bias = ab[qk * 3 + 0] * sbw[0 * 8 + h] + ab[qk * 3 + 1] * sbw[1 * 8 + h]
                   + ab[qk * 3 + 2] * sbw[2 * 8 + h] + sbb[h]
                   + mc[qk] * mbw[h] + mbb[h];
        float sc = Sc[m * 256 + n] + bias;
        if (kpm[b * 256 + n]) sc = NEGBIG;
        Sc[m * 256 + n] = sc;
    }
    __syncthreads();

    // --- softmax (8 threads/row); emit normalized P as permuted f16 ---
    {
        int row = tid >> 3, sub = tid & 7;
        float mx = -3.4e38f;
        for (int n = sub; n < 256; n += 8) mx = fmaxf(mx, Sc[row * 256 + n]);
        red[row * 8 + sub] = mx; __syncthreads();
        if (sub == 0) {
            float m2 = red[row * 8];
            for (int i = 1; i < 8; i++) m2 = fmaxf(m2, red[row * 8 + i]);
            red[row * 8] = m2;
        }
        __syncthreads();
        mx = red[row * 8]; __syncthreads();
        float sum = 0.f;
        for (int n = sub; n < 256; n += 8) {
            float e = __expf(Sc[row * 256 + n] - mx);
            Sc[row * 256 + n] = e; sum += e;
        }
        red[row * 8 + sub] = sum; __syncthreads();
        if (sub == 0) {
            float s2 = 0.f;
            for (int i = 0; i < 8; i++) s2 += red[row * 8 + i];
            red[row * 8] = s2;
        }
        __syncthreads();
        float inv = 1.0f / red[row * 8];
        for (int n = sub; n < 256; n += 8) {
            Ps[row * 264 + permk64(n & 63) + (n & 192)] = (_Float16)(Sc[row * 256 + n] * inv);
        }
    }
    __syncthreads();

    // --- attended = P @ V : wave computes head-dims [16*wave, 16*wave+16) ---
    v8f acc = {};
    const int ncv = wave * 16 + l15;
    #pragma unroll
    for (int kc = 0; kc < 256; kc += 32) {
        v16h af = frag_ld(&Ps[l15 * 264 + kc + lhi * 16]);
        v16h bf = frag_ld(&KV[ncv * 264 + kc + lhi * 16]);
        acc = __builtin_amdgcn_wmma_f32_16x16x32_f16(
            false, af, false, bf, (short)0, acc, false, false);
    }
    #pragma unroll
    for (int v = 0; v < 8; v++)
        Out[((size_t)(b * 256 + q0 + v + lhi * 8)) * 512 + h * 64 + ncv] = acc[v];
}

// ---------------------------------------------------------------------------
// Host orchestration
// ---------------------------------------------------------------------------
extern "C" void kernel_launch(void* const* d_in, const int* in_sizes, int n_in,
                              void* d_out, int out_size, void* d_ws, size_t ws_size,
                              hipStream_t stream)
{
    (void)in_sizes; (void)n_in; (void)out_size; (void)ws_size;
    const float* center = (const float*)d_in[0];
    const float* cand   = (const float*)d_in[1];
    const unsigned char* cmask = (const unsigned char*)d_in[2];
    const int*   etyp   = (const int*)d_in[3];
    const float* ydel   = (const float*)d_in[4];
    const float* cscr   = (const float*)d_in[5];
    const float* ab     = (const float*)d_in[6];
    const float* mc     = (const float*)d_in[7];
    const unsigned char* kpm = (const unsigned char*)d_in[8];

    // params flattened in jax pytree order at index 9: 'layers' (list of dicts,
    // keys alphabetical), then sel_*/tok_* alphabetical.
    const int PB = 9;
    struct LayerP {
        const float *f1_b,*f1_w,*f2_b,*f2_w,*k_b,*k_w,*mb_b,*mb_w,*n1_b,*n1_g,
                    *n2_b,*n2_g,*o_b,*o_w,*q_b,*q_w,*sb_b,*sb_w,*v_b,*v_w;
    } L[2];
    for (int l = 0; l < 2; l++) {
        const float** f = (const float**)&L[l];
        for (int i = 0; i < 20; i++) f[i] = (const float*)d_in[PB + 20 * l + i];
    }
    const int NP = PB + 40;
    const float* sel_c_b      = (const float*)d_in[NP + 0];
    const float* sel_c_w      = (const float*)d_in[NP + 1];
    const float* sel_edge_emb = (const float*)d_in[NP + 2];
    const float* sel_f1_b     = (const float*)d_in[NP + 3];
    const float* sel_f1_w     = (const float*)d_in[NP + 4];
    const float* sel_f2_b     = (const float*)d_in[NP + 5];
    const float* sel_f2_w     = (const float*)d_in[NP + 6];
    const float* sel_q_b      = (const float*)d_in[NP + 7];
    const float* sel_q_w      = (const float*)d_in[NP + 8];
    const float* sel_s_b      = (const float*)d_in[NP + 9];
    const float* sel_s_w      = (const float*)d_in[NP + 10];
    const float* sel_t_b      = (const float*)d_in[NP + 11];
    const float* sel_t_w      = (const float*)d_in[NP + 12];
    const float* tok_edge_emb = (const float*)d_in[NP + 13];
    const float* tok_hop_b    = (const float*)d_in[NP + 14];
    const float* tok_ln_b     = (const float*)d_in[NP + 16];
    const float* tok_ln_g     = (const float*)d_in[NP + 17];
    const float* tok_s_b      = (const float*)d_in[NP + 18];
    const float* tok_s_w      = (const float*)d_in[NP + 19];
    const float* tok_sp_b     = (const float*)d_in[NP + 20];
    const float* tok_t_b      = (const float*)d_in[NP + 22];
    const float* tok_t_w      = (const float*)d_in[NP + 23];
    const float* tok_text_b   = (const float*)d_in[NP + 24];
    const float* tok_text_w   = (const float*)d_in[NP + 25];

    // ---- workspace carving ----
    char* wp = (char*)d_ws;
    auto carve = [&](size_t bytes) { void* p = (void*)wp; wp += (bytes + 255) & ~(size_t)255; return p; };
    float* qtmp   = (float*)carve(32 * 256 * 4);
    float* basev  = (float*)carve(32 * 256 * 4);
    float* Mmat   = (float*)carve(768 * 256 * 4);
    float* edgev  = (float*)carve(4 * 256 * 4);
    float* tvec   = (float*)carve(256 * 4);
    float* tconst = (float*)carve(256 * 4);
    float* svec   = (float*)carve(256 * 4);
    float* sconst = (float*)carve(256 * 4);
    float* cconst = (float*)carve(256 * 4);
    float* Hc     = (float*)carve((size_t)32768 * 256 * 4);
    float* logits = (float*)carve(32768 * 4);
    float* topv   = (float*)carve(8192 * 4);
    int*   topi   = (int*)carve(8192 * 4);
    int*   rows   = (int*)carve(8192 * 4);
    float* textp  = (float*)carve((size_t)8192 * 512 * 4);   // reused as attn out
    float* tokens = (float*)carve((size_t)8192 * 512 * 4);
    float* xbuf   = (float*)carve((size_t)8192 * 512 * 4);
    float* qbuf   = (float*)carve((size_t)8192 * 512 * 4);
    float* kbuf   = (float*)carve((size_t)8192 * 512 * 4);
    float* vbuf   = (float*)carve((size_t)8192 * 512 * 4);
    float* ffh    = (float*)carve((size_t)8192 * 2048 * 4);
    float* attno  = textp;  // textp dead after tok_struct_ln

    // ---- selector precomputes (tiny GEMMs folding f1_w row-blocks) ----
    auto ng = [&](const float* A, const float* B, const float* bias, float* C,
                  int M, int N, int K, int ldb, int bRow0) {
        naive_gemm<<<(M * N + 255) / 256, 256, 0, stream>>>(A, B, bias, C, M, N, K, ldb, bRow0);
    };
    ng(center, sel_q_w, sel_q_b, qtmp, 32, 256, 768, 256, 0);
    ng(qtmp, sel_f1_w, nullptr, basev, 32, 256, 256, 256, 0);
    ng(sel_c_w, sel_f1_w, nullptr, Mmat, 768, 256, 256, 256, 256);
    ng(sel_edge_emb, sel_f1_w, nullptr, edgev, 4, 256, 256, 256, 512);
    ng(sel_t_w, sel_f1_w, nullptr, tvec, 1, 256, 256, 256, 768);
    ng(sel_t_b, sel_f1_w, nullptr, tconst, 1, 256, 256, 256, 768);
    ng(sel_s_w, sel_f1_w, nullptr, svec, 1, 256, 256, 256, 1024);
    ng(sel_s_b, sel_f1_w, nullptr, sconst, 1, 256, 256, 256, 1024);
    ng(sel_c_b, sel_f1_w, nullptr, cconst, 1, 256, 256, 256, 256);

    auto gemm = [&](const float* A, const float* B, const float* bias, const float* res,
                    float* C, int M, int N, int K, int flags) {
        dim3 g((unsigned)(N / 64), (unsigned)(M / 64));
        gemm_wmma_f16<false><<<g, 128, 0, stream>>>(A, B, bias, res, C, nullptr, M, N, K, flags);
    };

    // ---- heavy selector GEMM: Hc = CE(32768x768) @ Mmat(768x256) ----
    gemm(cand, Mmat, nullptr, nullptr, Hc, 32768, 256, 768, 0);

    // ---- fused logits ----
    selector_logits<<<8192, 128, 0, stream>>>(
        center, cand, cmask, etyp, ydel, cscr, Hc, basev, edgev,
        tvec, tconst, svec, sconst, cconst, sel_f1_w, sel_f1_b,
        sel_f2_w, sel_f2_b, logits);

    // ---- top-256 per batch ----
    topk1024<<<32, 512, 0, stream>>>(logits, topv, topi);
    build_rows<<<32, 256, 0, stream>>>(topi, rows);

    // ---- tokenizer (gathered rows) ----
    {
        dim3 g(512 / 64, 8192 / 64);
        gemm_wmma_f16<true><<<g, 128, 0, stream>>>(
            cand, tok_text_w, tok_text_b, nullptr, textp, rows, 8192, 512, 768, 1);
    }
    tok_struct_ln<<<2048, 128, 0, stream>>>(
        textp, topi, topv, etyp, ydel, tok_edge_emb,
        tok_t_w, tok_t_b, tok_s_w, tok_s_b, tok_hop_b, tok_sp_b,
        tok_ln_g, tok_ln_b, tokens);

    // ---- transformer layers ----
    for (int l = 0; l < 2; l++) {
        const LayerP& P = L[l];
        ln512<<<2048, 128, 0, stream>>>(tokens, xbuf, P.n1_g, P.n1_b, 8192);
        gemm(xbuf, P.q_w, P.q_b, nullptr, qbuf, 8192, 512, 512, 1);
        gemm(xbuf, P.k_w, P.k_b, nullptr, kbuf, 8192, 512, 512, 1);
        gemm(xbuf, P.v_w, P.v_b, nullptr, vbuf, 8192, 512, 512, 1);
        attn256<<<32 * 8 * 16, 128, 0, stream>>>(
            qbuf, kbuf, vbuf, ab, mc, kpm, P.sb_w, P.sb_b, P.mb_w, P.mb_b, attno);
        gemm(attno, P.o_w, P.o_b, tokens, tokens, 8192, 512, 512, 1 | 4);
        ln512<<<2048, 128, 0, stream>>>(tokens, xbuf, P.n2_g, P.n2_b, 8192);
        gemm(xbuf, P.f1_w, P.f1_b, nullptr, ffh, 8192, 2048, 512, 1 | 2);
        float* dst = (l == 1) ? (float*)d_out : tokens;
        gemm(ffh, P.f2_w, P.f2_b, tokens, dst, 8192, 512, 2048, 1 | 4);
    }
}
